// LSTM_2946347565336
// MI455X (gfx1250) — matrix operand order repaired and checked by reading, not compiled, not run on hardware
//
#include <hip/hip_runtime.h>
#include <hip/hip_bf16.h>
#include <cstdint>

// ---------------------------------------------------------------------------
// Problem dims (match the JAX reference)
// ---------------------------------------------------------------------------
#define B_ 64
#define T_ 1024
#define E_ 768
#define H_ 256
#define V_ 3072

typedef __bf16 bf16_t;
typedef __attribute__((ext_vector_type(16))) __bf16   v16bf;
typedef __attribute__((ext_vector_type(8)))  float    v8f;
typedef __attribute__((ext_vector_type(4)))  uint32_t v4u;

union Frag32B { v16bf bf; v4u q[2]; };

__device__ __forceinline__ v8f zero_v8f() {
    v8f z = {0.f, 0.f, 0.f, 0.f, 0.f, 0.f, 0.f, 0.f};
    return z;
}

// A-matrix 16x32 bf16 fragment (ISA 7.12.2): lane L -> row m = L%16,
// K-half kh = (L<16)?0:8 ; VGPR0..3 hold K=kh..kh+7, VGPR4..7 hold K=kh+16..kh+23.
__device__ __forceinline__ v16bf load_frag_a(const bf16_t* base, int ld, int k0, int lane) {
    int m  = lane & 15;
    int kh = (lane < 16) ? 0 : 8;
    const bf16_t* p = base + (size_t)m * ld + k0 + kh;
    Frag32B f;
    f.q[0] = *(const v4u*)(p);        // K = k0+kh   .. k0+kh+7
    f.q[1] = *(const v4u*)(p + 16);   // K = k0+kh+16.. k0+kh+23
    return f.bf;
}

// B-matrix 32x16 bf16 fragment: lane L -> col n = L%16, kb=(L<16)?0:8; K=kb..kb+7, kb+16..kb+23.
// baseT is TRANSPOSED weight [N][K] row-major, so per-lane K-values contiguous.
__device__ __forceinline__ v16bf load_frag_b(const bf16_t* baseT, int ld, int k0,
                                             int n0, int lane) {
    int n  = lane & 15;
    int kb = (lane < 16) ? 0 : 8;
    const bf16_t* p = baseT + (size_t)(n0 + n) * ld + k0 + kb;
    Frag32B f;
    f.q[0] = *(const v4u*)(p);
    f.q[1] = *(const v4u*)(p + 16);
    return f.bf;
}

__device__ __forceinline__ v8f wmma_bf16(v16bf a, v16bf b, v8f c) {
    return __builtin_amdgcn_wmma_f32_16x16x32_bf16(false, a, false, b,
                                                   (short)0, c, false, false);
}

// fast tanh via exp: tanh(x) = (1 - e^2)/(1 + e^2) with e = exp(-x), clamped
__device__ __forceinline__ float fast_tanh_from_e(float e) {
    float e2 = e * e;
    return (1.0f - e2) / (1.0f + e2);
}

// ---------------------------------------------------------------------------
// Kernel 0: cast emb -> bf16, transpose+cast Wxh, Whh, Wch, Wy -> bf16 [N][K]
// ---------------------------------------------------------------------------
__global__ void prep_kernel(const float* __restrict__ emb,
                            const float* __restrict__ Wxh,
                            const float* __restrict__ Whh,
                            const float* __restrict__ Wch,
                            const float* __restrict__ Wy,
                            bf16_t* __restrict__ emb_bf,
                            bf16_t* __restrict__ WxhT,
                            bf16_t* __restrict__ WhhT,
                            bf16_t* __restrict__ WchT,
                            bf16_t* __restrict__ WyT) {
    const int N_EMB = V_ * E_;
    const int N_WXH = E_ * H_;
    const int N_WHH = H_ * H_;
    const int N_WY  = H_ * V_;
    const int total = N_EMB + N_WXH + 2 * N_WHH + N_WY;
    for (int i = blockIdx.x * blockDim.x + threadIdx.x; i < total;
         i += gridDim.x * blockDim.x) {
        int j = i;
        if (j < N_EMB) { emb_bf[j] = (bf16_t)emb[j]; continue; }
        j -= N_EMB;
        if (j < N_WXH) {                     // WxhT[n][e] = Wxh[e][n]
            int n = j / E_, e = j % E_;
            WxhT[j] = (bf16_t)Wxh[(size_t)e * H_ + n];
            continue;
        }
        j -= N_WXH;
        if (j < N_WHH) {                     // WhhT[n][k] = Whh[k][n]
            int n = j / H_, k = j % H_;
            WhhT[j] = (bf16_t)Whh[(size_t)k * H_ + n];
            continue;
        }
        j -= N_WHH;
        if (j < N_WHH) {                     // WchT[n][k] = Wch[k][n]
            int n = j / H_, k = j % H_;
            WchT[j] = (bf16_t)Wch[(size_t)k * H_ + n];
            continue;
        }
        j -= N_WHH;
        {                                    // WyT[v][k] = Wy[k][v]
            int v = j / H_, k = j % H_;
            WyT[j] = (bf16_t)Wy[(size_t)k * V_ + v];
        }
    }
}

// ---------------------------------------------------------------------------
// Kernel 1: xW[t][b][h] = emb[idx[b,t]] @ Wxh  (gathered bf16 GEMM, f32 acc)
// Software-pipelined: fragments for k+1 are loaded into FRESH registers
// before the k-step WMMAs issue, so s_wait_loadcnt overlaps with math.
// ---------------------------------------------------------------------------
__global__ __launch_bounds__(256) void gather_xw_kernel(
    const int* __restrict__ idx, const bf16_t* __restrict__ emb_bf,
    const bf16_t* __restrict__ WxhT, float* __restrict__ xW) {
    const int lane = threadIdx.x & 31;
    const int wave = threadIdx.x >> 5;
    const int job  = blockIdx.x * 8 + wave;   // 0..16383
    const int mtile = job >> 2;               // 0..4095
    const int ng    = job & 3;                // 0..3 (64-col group)
    const int r0 = mtile * 16;

    const int m = lane & 15;
    const int r = r0 + m;
    const int t = r >> 6;
    const int b = r & 63;
    const int erow = idx[b * T_ + t];
    const int kh = (lane < 16) ? 0 : 8;
    const int kb = (lane < 16) ? 0 : 8;
    const bf16_t* arow = emb_bf + (size_t)erow * E_ + kh;       // per-lane A ptr
    const bf16_t* brow[4];
#pragma unroll
    for (int j = 0; j < 4; ++j)
        brow[j] = WxhT + (size_t)(ng * 64 + j * 16 + (lane & 15)) * E_ + kb;

    v8f acc[4] = {zero_v8f(), zero_v8f(), zero_v8f(), zero_v8f()};

    Frag32B fa, fb[4];
    fa.q[0] = *(const v4u*)(arow);
    fa.q[1] = *(const v4u*)(arow + 16);
#pragma unroll
    for (int j = 0; j < 4; ++j) {
        fb[j].q[0] = *(const v4u*)(brow[j]);
        fb[j].q[1] = *(const v4u*)(brow[j] + 16);
    }

#pragma unroll
    for (int k0 = 0; k0 < E_; k0 += 32) {
        Frag32B fa_n, fb_n[4];
        if (k0 + 32 < E_) {                   // prefetch next k-step
            const bf16_t* pa = arow + k0 + 32;
            fa_n.q[0] = *(const v4u*)(pa);
            fa_n.q[1] = *(const v4u*)(pa + 16);
#pragma unroll
            for (int j = 0; j < 4; ++j) {
                const bf16_t* pb = brow[j] + k0 + 32;
                fb_n[j].q[0] = *(const v4u*)(pb);
                fb_n[j].q[1] = *(const v4u*)(pb + 16);
            }
        }
#pragma unroll
        for (int j = 0; j < 4; ++j)
            acc[j] = wmma_bf16(fa.bf, fb[j].bf, acc[j]);
        fa = fa_n;
#pragma unroll
        for (int j = 0; j < 4; ++j) fb[j] = fb_n[j];
    }

    const int mh = (lane < 16) ? 0 : 8;
    const int n  = lane & 15;
#pragma unroll
    for (int j = 0; j < 4; ++j) {
        float* col = xW + (size_t)(r0 + mh) * H_ + ng * 64 + j * 16 + n;
#pragma unroll
        for (int jj = 0; jj < 8; ++jj) col[(size_t)jj * H_] = acc[j][jj];
    }
}

// ---------------------------------------------------------------------------
// Kernel 2: persistent single-workgroup recurrent scan on one WGP.
// LDS: h,c (bf16, 64 KB) + WhhT,WchT staged once (256 KB) = 320 KB, the full
// CDNA5 per-workgroup LDS budget -> every per-step operand is a ds_load.
// ---------------------------------------------------------------------------
__global__ __launch_bounds__(1024) void lstm_scan_kernel(
    const float* __restrict__ xW, const bf16_t* __restrict__ WhhT,
    const bf16_t* __restrict__ WchT, const float* __restrict__ Bh,
    const float* __restrict__ Bc, float* __restrict__ h_out,
    float* __restrict__ c_out, bf16_t* __restrict__ h_bf_out) {
    __shared__ __align__(16) bf16_t h_bf[B_][H_];      //  32 KB
    __shared__ __align__(16) bf16_t c_bf[B_][H_];      //  32 KB
    __shared__ __align__(16) bf16_t whh_s[H_][H_];     // 128 KB
    __shared__ __align__(16) bf16_t wch_s[H_][H_];     // 128 KB

    // ---- one-time staging: weights -> LDS, h/c -> 0 ----
    {
        const v4u* s1 = (const v4u*)WhhT;  v4u* d1 = (v4u*)&whh_s[0][0];
        const v4u* s2 = (const v4u*)WchT;  v4u* d2 = (v4u*)&wch_s[0][0];
        const int nvec = (H_ * H_) / 8;    // 8 bf16 per 16B
        for (int i = threadIdx.x; i < nvec; i += 1024) d1[i] = s1[i];
        for (int i = threadIdx.x; i < nvec; i += 1024) d2[i] = s2[i];
        for (int i = threadIdx.x; i < B_ * H_; i += 1024) {
            (&h_bf[0][0])[i] = (bf16_t)0.0f;
            (&c_bf[0][0])[i] = (bf16_t)0.0f;
        }
    }
    __syncthreads();

    const int lane = threadIdx.x & 31;
    const int wave = threadIdx.x >> 5;       // 0..31
    const int n    = lane & 15;
    const int mh   = (lane < 16) ? 0 : 8;

    for (int t = 0; t < T_; ++t) {
        if (t + 1 < T_)   // pull next timestep's 64 KB xW slab toward the WGP
            __builtin_prefetch(xW + (size_t)(t + 1) * B_ * H_ + threadIdx.x * 16, 0, 1);

        float hn_reg[2][8], cn_reg[2][8];
#pragma unroll
        for (int p = 0; p < 2; ++p) {
            const int pair = wave * 2 + p;   // 0..63
            const int m0   = (pair >> 4) * 16;
            const int n0   = (pair & 15) * 16;
            v8f acc_pre  = zero_v8f();
            v8f acc_prev = zero_v8f();
#pragma unroll
            for (int k0 = 0; k0 < H_; k0 += 32) {
                v16bf ah = load_frag_a(&h_bf[m0][0], H_, k0, lane);
                v16bf bh = load_frag_b(&whh_s[0][0], H_, k0, n0, lane);
                acc_pre  = wmma_bf16(ah, bh, acc_pre);
                v16bf ac = load_frag_a(&c_bf[m0][0], H_, k0, lane);
                v16bf bc = load_frag_b(&wch_s[0][0], H_, k0, n0, lane);
                acc_prev = wmma_bf16(ac, bc, acc_prev);
            }
            const int nn  = n0 + n;
            const float bhv = Bh[nn];
            const float bcv = Bc[nn];
            const float* xwcol = xW + (size_t)t * B_ * H_ + (size_t)(m0 + mh) * H_ + nn;
#pragma unroll
            for (int j = 0; j < 8; ++j) {
                float pre  = acc_pre[j] + xwcol[(size_t)j * H_] + bhv;
                pre = fminf(fmaxf(pre, -30.0f), 30.0f);
                float prev = acc_prev[j] + bcv;
                float e1 = __expf(-pre);
                float s  = 1.0f / (1.0f + e1);              // sigmoid(pre)
                float tp = fast_tanh_from_e(e1);            // tanh(pre)
                float cn = s * (tp + prev);
                float cc = fminf(fmaxf(cn, -30.0f), 30.0f);
                float hn = fast_tanh_from_e(__expf(-cc)) * s;
                hn_reg[p][j] = hn;
                cn_reg[p][j] = cn;
            }
        }
        __syncthreads();   // all waves done READING old h/c
#pragma unroll
        for (int p = 0; p < 2; ++p) {
            const int pair = wave * 2 + p;
            const int m0   = (pair >> 4) * 16;
            const int nn   = (pair & 15) * 16 + n;
#pragma unroll
            for (int j = 0; j < 8; ++j) {
                const int bm = m0 + mh + j;
                h_bf[bm][nn] = (bf16_t)hn_reg[p][j];
                c_bf[bm][nn] = (bf16_t)cn_reg[p][j];
                if (t == T_ - 1) {
                    h_out[(size_t)bm * H_ + nn]    = hn_reg[p][j];
                    c_out[(size_t)bm * H_ + nn]    = cn_reg[p][j];
                    h_bf_out[(size_t)bm * H_ + nn] = (bf16_t)hn_reg[p][j];
                }
            }
        }
        __syncthreads();   // new h/c visible before next step reads
    }
}

// ---------------------------------------------------------------------------
// Kernel 3: out[b][v] = h @ Wy + By   (64 x 3072, K=256). Pipelined k-loop.
// ---------------------------------------------------------------------------
__global__ __launch_bounds__(256) void logits_kernel(
    const bf16_t* __restrict__ h_bf, const bf16_t* __restrict__ WyT,
    const float* __restrict__ By, float* __restrict__ out) {
    const int lane = threadIdx.x & 31;
    const int wave = threadIdx.x >> 5;
    const int job  = blockIdx.x * 8 + wave;  // 0..191 = mtile(4) x ngroup(48)
    const int mtile = job / 48;
    const int ng    = job % 48;
    const int m0 = mtile * 16;

    const int kh = (lane < 16) ? 0 : 8;
    const int kb = (lane < 16) ? 0 : 8;
    const bf16_t* arow = h_bf + (size_t)(m0 + (lane & 15)) * H_ + kh;
    const bf16_t* brow[4];
#pragma unroll
    for (int j = 0; j < 4; ++j)
        brow[j] = WyT + (size_t)(ng * 64 + j * 16 + (lane & 15)) * H_ + kb;

    v8f acc[4] = {zero_v8f(), zero_v8f(), zero_v8f(), zero_v8f()};
    Frag32B fa, fb[4];
    fa.q[0] = *(const v4u*)(arow);
    fa.q[1] = *(const v4u*)(arow + 16);
#pragma unroll
    for (int j = 0; j < 4; ++j) {
        fb[j].q[0] = *(const v4u*)(brow[j]);
        fb[j].q[1] = *(const v4u*)(brow[j] + 16);
    }
#pragma unroll
    for (int k0 = 0; k0 < H_; k0 += 32) {
        Frag32B fa_n, fb_n[4];
        if (k0 + 32 < H_) {
            const bf16_t* pa = arow + k0 + 32;
            fa_n.q[0] = *(const v4u*)(pa);
            fa_n.q[1] = *(const v4u*)(pa + 16);
#pragma unroll
            for (int j = 0; j < 4; ++j) {
                const bf16_t* pb = brow[j] + k0 + 32;
                fb_n[j].q[0] = *(const v4u*)(pb);
                fb_n[j].q[1] = *(const v4u*)(pb + 16);
            }
        }
#pragma unroll
        for (int j = 0; j < 4; ++j)
            acc[j] = wmma_bf16(fa.bf, fb[j].bf, acc[j]);
        fa = fa_n;
#pragma unroll
        for (int j = 0; j < 4; ++j) fb[j] = fb_n[j];
    }

    const int mh = (lane < 16) ? 0 : 8;
    const int n  = lane & 15;
#pragma unroll
    for (int j = 0; j < 4; ++j) {
        const int nn = ng * 64 + j * 16 + n;
        const float bias = By[nn];
#pragma unroll
        for (int jj = 0; jj < 8; ++jj)
            out[(size_t)(m0 + mh + jj) * V_ + nn] = acc[j][jj] + bias;
    }
}

// ---------------------------------------------------------------------------
// Host entry
// ---------------------------------------------------------------------------
extern "C" void kernel_launch(void* const* d_in, const int* in_sizes, int n_in,
                              void* d_out, int out_size, void* d_ws, size_t ws_size,
                              hipStream_t stream) {
    const int*   idx = (const int*)d_in[0];
    const float* emb = (const float*)d_in[1];
    const float* Wxh = (const float*)d_in[2];
    const float* Whh = (const float*)d_in[3];
    const float* Wch = (const float*)d_in[4];
    const float* Wy  = (const float*)d_in[5];
    const float* By  = (const float*)d_in[6];
    const float* Bh  = (const float*)d_in[7];
    const float* Bc  = (const float*)d_in[8];

    char* ws = (char*)d_ws;
    size_t off = 0;
    float*  xW     = (float*)(ws + off);  off += (size_t)T_ * B_ * H_ * 4;  // 64 MB
    bf16_t* emb_bf = (bf16_t*)(ws + off); off += (size_t)V_ * E_ * 2;       // 4.7 MB
    bf16_t* WxhT   = (bf16_t*)(ws + off); off += (size_t)H_ * E_ * 2;
    bf16_t* WhhT   = (bf16_t*)(ws + off); off += (size_t)H_ * H_ * 2;
    bf16_t* WchT   = (bf16_t*)(ws + off); off += (size_t)H_ * H_ * 2;
    bf16_t* WyT    = (bf16_t*)(ws + off); off += (size_t)V_ * H_ * 2;
    bf16_t* h_bf_g = (bf16_t*)(ws + off); off += (size_t)B_ * H_ * 2;

    float* out   = (float*)d_out;                 // logits [B,V]
    float* h_out = out + (size_t)B_ * V_;         // h [B,H]
    float* c_out = h_out + (size_t)B_ * H_;       // c [B,H]

    prep_kernel<<<2048, 256, 0, stream>>>(emb, Wxh, Whh, Wch, Wy,
                                          emb_bf, WxhT, WhhT, WchT, WyT);
    gather_xw_kernel<<<2048, 256, 0, stream>>>(idx, emb_bf, WxhT, xW);
    lstm_scan_kernel<<<1, 1024, 0, stream>>>(xW, WhhT, WchT, Bh, Bc,
                                             h_out, c_out, h_bf_g);
    logits_kernel<<<24, 256, 0, stream>>>(h_bf_g, WyT, By, out);
}
